// ParallelMultiHeadAttention_4922032521390
// MI455X (gfx1250) — compile-verified, matrix operand order checked by reading, not run on hardware
//
#include <hip/hip_runtime.h>
#include <hip/hip_bf16.h>

// ---------------------------------------------------------------------------
// Causal multi-head attention forward for MI455X (gfx1250, wave32).
// B=4, T=2048, C=1024, H=16, D=64.
// All GEMM work runs on v_wmma_f32_16x16x32_bf16 (fp32 inputs truncated to
// bf16 once; f32 accumulation throughout).
// ---------------------------------------------------------------------------

#define AB 4          // batch
#define AT 2048       // sequence
#define AC 1024       // embed
#define AH 16         // heads
#define AD 64         // head dim
#define A3C 3072

typedef __bf16 bf16_t;
typedef __attribute__((ext_vector_type(16))) __bf16 bf16x16;
typedef __attribute__((ext_vector_type(8)))  __bf16 bf16x8;
typedef __attribute__((ext_vector_type(8)))  float  f32x8;

__device__ __forceinline__ f32x8 wmma_bf16(bf16x16 a, bf16x16 b, f32x8 c) {
  // D = A(16x32 bf16) * B(32x16 bf16) + C(16x16 f32)
  return __builtin_amdgcn_wmma_f32_16x16x32_bf16(
      /*neg_a=*/false, a, /*neg_b=*/false, b,
      /*c_mod=*/(short)0, c, /*reuse_a=*/false, /*reuse_b=*/false);
}

__device__ __forceinline__ bf16x16 make_frag(bf16x8 lo, bf16x8 hi) {
  bf16x16 f;
#pragma unroll
  for (int i = 0; i < 8; ++i) { f[i] = lo[i]; f[i + 8] = hi[i]; }
  return f;
}

// Load a 16-element A/B fragment slice for this lane: 8 bf16 at p, 8 at p+16.
__device__ __forceinline__ bf16x16 load_frag(const bf16_t* __restrict__ p) {
  return make_frag(*(const bf16x8*)p, *(const bf16x8*)(p + 16));
}

// ---------------------------------------------------------------------------
// Kernel 1a: fp32 -> bf16 elementwise
// ---------------------------------------------------------------------------
__global__ void cvt_bf16(const float* __restrict__ src, bf16_t* __restrict__ dst, int n) {
  int i = blockIdx.x * blockDim.x + threadIdx.x;
  int stride = gridDim.x * blockDim.x;
  for (; i < n; i += stride) dst[i] = (bf16_t)src[i];
}

// Kernel 1b: fp32 [K,N] -> bf16 [N,K] (transpose so GEMM B-frags load
// contiguously along K)
__global__ void cvt_wT(const float* __restrict__ w, bf16_t* __restrict__ wt,
                       int K, int N) {
  int i = blockIdx.x * blockDim.x + threadIdx.x;
  int stride = gridDim.x * blockDim.x;
  int total = K * N;
  for (; i < total; i += stride) {
    int nn = i / K;
    int kk = i - nn * K;
    wt[i] = (bf16_t)w[kk * N + nn];
  }
}

// ---------------------------------------------------------------------------
// Shared GEMM core: wave computes a 32x64 tile (2x4 WMMA tiles) of
// A[M,1024] * Bt[N,1024]^T, K-loop step 32.
// ---------------------------------------------------------------------------
__device__ __forceinline__ void gemm_core_32x64(
    const bf16_t* __restrict__ a, const bf16_t* __restrict__ bt,
    int m0, int n0, int lr, int koff, f32x8 (&acc)[2][4]) {
  for (int k0 = 0; k0 < AC; k0 += 32) {
    bf16x16 afrag[2], bfrag[4];
#pragma unroll
    for (int rt = 0; rt < 2; ++rt) {
      const bf16_t* p = a + (size_t)(m0 + rt * 16 + lr) * AC + k0 + koff;
      __builtin_prefetch(p + 64, 0, 1);     // global_prefetch_b8: next K blocks
      afrag[rt] = load_frag(p);
    }
#pragma unroll
    for (int ct = 0; ct < 4; ++ct) {
      const bf16_t* p = bt + (size_t)(n0 + ct * 16 + lr) * AC + k0 + koff;
      bfrag[ct] = load_frag(p);
    }
#pragma unroll
    for (int rt = 0; rt < 2; ++rt)
#pragma unroll
      for (int ct = 0; ct < 4; ++ct)
        acc[rt][ct] = wmma_bf16(afrag[rt], bfrag[ct], acc[rt][ct]);
  }
}

// ---------------------------------------------------------------------------
// Kernel 2: QKV GEMM.  xb[M=8192, K=1024] * wqkvT[N=3072, K=1024]^T.
// Block: 256 thr = 8 waves (2 wave-rows x 4 wave-cols), block tile 64x256.
// blockIdx.y uniformly selects the Q / K / V region (region width 1024 is a
// multiple of the 256-wide block tile), so the scatter epilogue has NO
// per-element branches.  Q,K stored [B,H,T,D]; V stored transposed [B,H,D,T].
// ---------------------------------------------------------------------------
__global__ __launch_bounds__(256) void qkv_gemm(
    const bf16_t* __restrict__ xb, const bf16_t* __restrict__ wT,
    bf16_t* __restrict__ qo, bf16_t* __restrict__ ko, bf16_t* __restrict__ vT) {
  const int lane = threadIdx.x & 31;
  const int wave = threadIdx.x >> 5;
  const int m0 = blockIdx.x * 64 + (wave >> 2) * 32;
  const int n0 = blockIdx.y * 256 + (wave & 3) * 64;
  const int lr = lane & 15;
  const int lh = lane >> 4;
  const int koff = lh * 8;

  f32x8 acc[2][4] = {};
  gemm_core_32x64(xb, wT, m0, n0, lr, koff, acc);

  const int region = blockIdx.y >> 2;        // 0=Q, 1=K, 2=V (uniform)
  const int ncol0  = n0 - region * AC;       // column within the region

  if (region < 2) {                          // Q or K -> [B,H,T,D]
    bf16_t* __restrict__ dst = (region == 0) ? qo : ko;
#pragma unroll
    for (int rt = 0; rt < 2; ++rt) {
#pragma unroll
      for (int ct = 0; ct < 4; ++ct) {
#pragma unroll
        for (int i = 0; i < 8; ++i) {
          const int row = m0 + rt * 16 + i + (lh << 3);   // [0, B*T)
          const int nn  = ncol0 + ct * 16 + lr;           // [0, C)
          const int b = row >> 11, t = row & (AT - 1);
          const int h = nn >> 6,  d = nn & (AD - 1);
          dst[(((size_t)b * AH + h) * AT + t) * AD + d] = (bf16_t)acc[rt][ct][i];
        }
      }
    }
  } else {                                   // V -> transposed [B,H,D,T]
#pragma unroll
    for (int rt = 0; rt < 2; ++rt) {
#pragma unroll
      for (int ct = 0; ct < 4; ++ct) {
#pragma unroll
        for (int i = 0; i < 8; ++i) {
          const int row = m0 + rt * 16 + i + (lh << 3);
          const int nn  = ncol0 + ct * 16 + lr;
          const int b = row >> 11, t = row & (AT - 1);
          const int h = nn >> 6,  d = nn & (AD - 1);
          vT[(((size_t)b * AH + h) * AD + d) * AT + t] = (bf16_t)acc[rt][ct][i];
        }
      }
    }
  }
}

// ---------------------------------------------------------------------------
// Kernel 3: flash attention.  One wave per 16-query tile of one (b,h).
// Streams 32-key tiles with online softmax; QK^T and P*V on WMMA.
// The bulk of the stream is unmasked; exactly one diagonal tile gets the
// causal mask.  P re-layout C->A goes through a per-wave LDS staging tile.
// ---------------------------------------------------------------------------
struct AttnState {
  f32x8 o[4];
  float m_i[8];
  float l_i[8];
};

template <bool MASK>
__device__ __forceinline__ void attn_tile(
    const bf16_t* __restrict__ kp, const bf16_t* __restrict__ vp,
    bf16_t* __restrict__ pst,                      // this wave's LDS tile
    const bf16x16 (&aq)[2], AttnState& st,
    int j0, int q0, int lr, int lh, int koff) {
  // S = Q K^T (two 16-key column tiles)
  f32x8 s0 = {}, s1 = {};
  {
    const bf16_t* p0 = kp + (size_t)(j0 + lr) * AD;
    const bf16_t* p1 = kp + (size_t)(j0 + 16 + lr) * AD;
    bf16x16 b00 = make_frag(*(const bf16x8*)(p0 + koff),      *(const bf16x8*)(p0 + 16 + koff));
    bf16x16 b01 = make_frag(*(const bf16x8*)(p0 + 32 + koff), *(const bf16x8*)(p0 + 48 + koff));
    bf16x16 b10 = make_frag(*(const bf16x8*)(p1 + koff),      *(const bf16x8*)(p1 + 16 + koff));
    bf16x16 b11 = make_frag(*(const bf16x8*)(p1 + 32 + koff), *(const bf16x8*)(p1 + 48 + koff));
    s0 = wmma_bf16(aq[0], b00, s0);
    s0 = wmma_bf16(aq[1], b01, s0);
    s1 = wmma_bf16(aq[0], b10, s1);
    s1 = wmma_bf16(aq[1], b11, s1);
  }

  // scale (+ mask) + online softmax; row stats align with C-layout slots
  const float sc = 0.125f;   // 1/sqrt(64)
  float p0v[8], p1v[8];
#pragma unroll
  for (int i = 0; i < 8; ++i) {
    float a = s0[i] * sc;
    float c = s1[i] * sc;
    if (MASK) {
      const int row = q0 + i + (lh << 3);
      if (j0 + lr > row)       a = -__builtin_inff();
      if (j0 + 16 + lr > row)  c = -__builtin_inff();
    }
    float mt = fmaxf(a, c);
#pragma unroll
    for (int off = 1; off < 16; off <<= 1)
      mt = fmaxf(mt, __shfl_xor(mt, off, 32));
    const float mnew = fmaxf(st.m_i[i], mt);
    const float e0 = __expf(a - mnew);
    const float e1 = __expf(c - mnew);
    p0v[i] = e0; p1v[i] = e1;
    float rs = e0 + e1;
#pragma unroll
    for (int off = 1; off < 16; off <<= 1)
      rs += __shfl_xor(rs, off, 32);
    const float corr = __expf(st.m_i[i] - mnew);
    st.l_i[i] = st.l_i[i] * corr + rs;
    st.m_i[i] = mnew;
#pragma unroll
    for (int n = 0; n < 4; ++n) st.o[n][i] *= corr;
  }

  // C-layout -> A-layout for P via per-wave LDS tile (same-wave DS in-order)
#pragma unroll
  for (int i = 0; i < 8; ++i) {
    const int row = i + (lh << 3);
    pst[row * 32 + lr]      = (bf16_t)p0v[i];
    pst[row * 32 + 16 + lr] = (bf16_t)p1v[i];
  }
  asm volatile("s_wait_dscnt 0" ::: "memory");
  bf16x16 pa;
  {
    const bf16_t* p = pst + lr * 32 + koff;
    pa = make_frag(*(const bf16x8*)p, *(const bf16x8*)(p + 16));
  }

  // O += P * V   (V pre-transposed: row d, contiguous keys)
#pragma unroll
  for (int n = 0; n < 4; ++n) {
    const bf16_t* p = vp + (size_t)(n * 16 + lr) * AT + j0 + koff;
    bf16x16 bv = make_frag(*(const bf16x8*)p, *(const bf16x8*)(p + 16));
    st.o[n] = wmma_bf16(pa, bv, st.o[n]);
  }
}

__global__ __launch_bounds__(128) void flash_attn(
    const bf16_t* __restrict__ q, const bf16_t* __restrict__ k,
    const bf16_t* __restrict__ vT, bf16_t* __restrict__ attn) {
  const int lane = threadIdx.x & 31;
  const int wave = threadIdx.x >> 5;                 // 0..3
  const int gw = blockIdx.x * 4 + wave;              // 0..8191
  const int qt = gw & 127;                           // q tile
  const int h  = (gw >> 7) & (AH - 1);
  const int b  = gw >> 11;
  const int q0 = qt * 16;
  const int lr = lane & 15;
  const int lh = lane >> 4;
  const int koff = lh * 8;

  const size_t bh = (size_t)b * AH + h;
  const bf16_t* qp = q  + bh * AT * AD;
  const bf16_t* kp = k  + bh * AT * AD;
  const bf16_t* vp = vT + bh * AD * AT;

  __shared__ __attribute__((aligned(16))) bf16_t pstage[4][16 * 32];
  bf16_t* pst = pstage[wave];

  // Q fragments: d = 0..31 and d = 32..63
  bf16x16 aq[2];
  {
    const bf16_t* p = qp + (size_t)(q0 + lr) * AD;
    aq[0] = make_frag(*(const bf16x8*)(p + koff),      *(const bf16x8*)(p + 16 + koff));
    aq[1] = make_frag(*(const bf16x8*)(p + 32 + koff), *(const bf16x8*)(p + 48 + koff));
  }

  AttnState st;
#pragma unroll
  for (int n = 0; n < 4; ++n) st.o[n] = (f32x8){};
#pragma unroll
  for (int i = 0; i < 8; ++i) { st.m_i[i] = -__builtin_inff(); st.l_i[i] = 0.f; }

  // Bulk: fully-unmasked key tiles (all keys <= q0-1 < every row in tile).
  const int nfull = q0 >> 5;
  for (int kt = 0; kt < nfull; ++kt)
    attn_tile<false>(kp, vp, pst, aq, st, kt * 32, q0, lr, lh, koff);
  // Exactly one diagonal tile needs the causal mask.
  attn_tile<true>(kp, vp, pst, aq, st, nfull * 32, q0, lr, lh, koff);

  // normalize and store bf16 [B,T,C] with head-major columns
#pragma unroll
  for (int i = 0; i < 8; ++i) {
    const float inv = 1.f / st.l_i[i];
    const int trow = q0 + i + (lh << 3);
    const size_t base = ((size_t)b * AT + trow) * AC + h * AD;
#pragma unroll
    for (int n = 0; n < 4; ++n)
      attn[base + n * 16 + lr] = (bf16_t)(st.o[n][i] * inv);
  }
}

// ---------------------------------------------------------------------------
// Kernel 4: projection GEMM + bias -> fp32 output.
// attn[M=8192, K=1024] * wprojT[N=1024, K=1024]^T + b_proj.
// ---------------------------------------------------------------------------
__global__ __launch_bounds__(256) void proj_gemm(
    const bf16_t* __restrict__ ab, const bf16_t* __restrict__ wT,
    const float* __restrict__ bias, float* __restrict__ out) {
  const int lane = threadIdx.x & 31;
  const int wave = threadIdx.x >> 5;
  const int m0 = blockIdx.x * 64 + (wave >> 2) * 32;
  const int n0 = blockIdx.y * 256 + (wave & 3) * 64;
  const int lr = lane & 15;
  const int lh = lane >> 4;
  const int koff = lh * 8;

  f32x8 acc[2][4] = {};
  gemm_core_32x64(ab, wT, m0, n0, lr, koff, acc);

#pragma unroll
  for (int rt = 0; rt < 2; ++rt) {
#pragma unroll
    for (int ct = 0; ct < 4; ++ct) {
      const int col = n0 + ct * 16 + lr;
      const float bv = bias[col];
#pragma unroll
      for (int i = 0; i < 8; ++i) {
        const int row = m0 + rt * 16 + i + (lh << 3);
        out[(size_t)row * AC + col] = acc[rt][ct][i] + bv;
      }
    }
  }
}

// ---------------------------------------------------------------------------
// Launcher
// ---------------------------------------------------------------------------
extern "C" void kernel_launch(void* const* d_in, const int* in_sizes, int n_in,
                              void* d_out, int out_size, void* d_ws, size_t ws_size,
                              hipStream_t stream) {
  (void)in_sizes; (void)n_in; (void)out_size; (void)ws_size;
  const float* x      = (const float*)d_in[0];   // [B,T,C]
  const float* w_qkv  = (const float*)d_in[1];   // [C,3C]
  const float* w_proj = (const float*)d_in[2];   // [C,C]
  const float* b_proj = (const float*)d_in[3];   // [C]
  float* out = (float*)d_out;

  char* ws = (char*)d_ws;
  const size_t MT = (size_t)AB * AT;             // 8192 rows
  bf16_t* xb    = (bf16_t*)ws;  ws += MT * AC * sizeof(bf16_t);          // 16 MB
  bf16_t* wqkvT = (bf16_t*)ws;  ws += (size_t)A3C * AC * sizeof(bf16_t); //  6 MB
  bf16_t* wprjT = (bf16_t*)ws;  ws += (size_t)AC * AC * sizeof(bf16_t);  //  2 MB
  bf16_t* qb    = (bf16_t*)ws;  ws += MT * AC * sizeof(bf16_t);          // 16 MB
  bf16_t* kb    = (bf16_t*)ws;  ws += MT * AC * sizeof(bf16_t);          // 16 MB
  bf16_t* vTb   = (bf16_t*)ws;  ws += MT * AC * sizeof(bf16_t);          // 16 MB
  bf16_t* atb   = (bf16_t*)ws;  ws += MT * AC * sizeof(bf16_t);          // 16 MB

  // 1) precision conversion (+ weight transposes)
  cvt_bf16<<<4096, 256, 0, stream>>>(x, xb, (int)(MT * AC));
  cvt_wT  <<<4096, 256, 0, stream>>>(w_qkv, wqkvT, AC, A3C);
  cvt_wT  <<<2048, 256, 0, stream>>>(w_proj, wprjT, AC, AC);

  // 2) QKV GEMM  (M/64 x N/256 blocks)
  qkv_gemm<<<dim3(MT / 64, A3C / 256), 256, 0, stream>>>(xb, wqkvT, qb, kb, vTb);

  // 3) flash attention: B*H*(T/16) = 8192 wave-tiles, 4 waves per block
  flash_attn<<<(AB * AH * (AT / 16)) / 4, 128, 0, stream>>>(qb, kb, vTb, atb);

  // 4) projection + bias
  proj_gemm<<<dim3(MT / 64, AC / 256), 256, 0, stream>>>(atb, wprjT, b_proj, out);
}
